// SovereignModel_77936476554067
// MI455X (gfx1250) — compile-verified, matrix operand order checked by reading, not run on hardware
//
#include <hip/hip_runtime.h>
#include <math.h>

// ---------------- model constants (from reference) ----------------
#define L_N   8
#define D_N   1024
#define H_N   16
#define HD_N  64
#define B_N   2
#define T_N   2048
#define V_N   32000
#define BT_N  (B_N * T_N)   // 4096 rows

// ---------------- WMMA types ----------------
typedef __bf16 v16bf __attribute__((ext_vector_type(16)));
typedef float  v8f   __attribute__((ext_vector_type(8)));

union BFrag {
    v16bf v;
    unsigned short s[16];
    uint4 u4[2];
};

__device__ __forceinline__ unsigned short f2bf(float f) {
    unsigned u = __float_as_uint(f);
    u += 0x7FFFu + ((u >> 16) & 1u);   // round-to-nearest-even
    return (unsigned short)(u >> 16);
}

__device__ __forceinline__ v8f wmma_bf16(const v16bf& a, const v16bf& b, const v8f& c) {
    return __builtin_amdgcn_wmma_f32_16x16x32_bf16(false, a, false, b, (short)0, c, false, false);
}

// ---------------- fp32 -> bf16 cast (weights) ----------------
__global__ void f32_to_bf16_k(const float* __restrict__ in, unsigned short* __restrict__ out, size_t n) {
    size_t i = (size_t)blockIdx.x * blockDim.x + threadIdx.x;
    size_t stride = (size_t)gridDim.x * blockDim.x;
    for (; i < n; i += stride) out[i] = f2bf(in[i]);
}

// ---------------- token + position embedding ----------------
__global__ __launch_bounds__(256) void embed_k(const int* __restrict__ idx,
                                               const float* __restrict__ wte,
                                               const float* __restrict__ wpe,
                                               float* __restrict__ x) {
    int row = blockIdx.x;            // b*T + t
    int t   = row % T_N;
    int tok = idx[row];
    const float* we = wte + (size_t)tok * D_N;
    const float* pe = wpe + (size_t)t * D_N;
    float* xr = x + (size_t)row * D_N;
#pragma unroll
    for (int j = 0; j < 4; ++j) {
        int c = threadIdx.x + j * 256;
        xr[c] = we[c] + pe[c];
    }
}

// ---------------- LayerNorm: fp32 in -> bf16 out ----------------
__global__ __launch_bounds__(256) void layernorm_bf16_k(const float* __restrict__ x,
                                                        const float* __restrict__ g,
                                                        const float* __restrict__ bta,
                                                        unsigned short* __restrict__ out) {
    __shared__ float red[256];
    int row = blockIdx.x;
    const float* xr = x + (size_t)row * D_N;
    float vals[4];
    float s = 0.0f;
#pragma unroll
    for (int j = 0; j < 4; ++j) { vals[j] = xr[threadIdx.x + j * 256]; s += vals[j]; }
    red[threadIdx.x] = s; __syncthreads();
    for (int o = 128; o > 0; o >>= 1) { if (threadIdx.x < o) red[threadIdx.x] += red[threadIdx.x + o]; __syncthreads(); }
    float mu = red[0] * (1.0f / D_N);
    __syncthreads();
    float v = 0.0f;
#pragma unroll
    for (int j = 0; j < 4; ++j) { float d = vals[j] - mu; v += d * d; }
    red[threadIdx.x] = v; __syncthreads();
    for (int o = 128; o > 0; o >>= 1) { if (threadIdx.x < o) red[threadIdx.x] += red[threadIdx.x + o]; __syncthreads(); }
    float rstd = rsqrtf(red[0] * (1.0f / D_N) + 1e-5f);
    unsigned short* orow = out + (size_t)row * D_N;
#pragma unroll
    for (int j = 0; j < 4; ++j) {
        int c = threadIdx.x + j * 256;
        orow[c] = f2bf((vals[j] - mu) * rstd * g[c] + bta[c]);
    }
}

// ---------------- generic bf16 WMMA GEMM:  C[M,N] = A[M,K] * W[N,K]^T (+bias,+gelu,+residual) ----
// block = 256 threads = 8 waves, 128x128 C tile.
// wave (w&3) -> 32-row strip (2 A frags), (w>>2) -> 64-col strip (4 B frags) => 8 WMMAs / K-step,
// 12 b128 loads / K-step = 1.5 loads per WMMA (vs 3.0 with a 16x32 wave tile).
__global__ __launch_bounds__(256) void gemm_bf16_wmma(const unsigned short* __restrict__ A,
                                                      const unsigned short* __restrict__ W,
                                                      const float* __restrict__ bias,
                                                      const float* __restrict__ residual,
                                                      float* __restrict__ outF,
                                                      unsigned short* __restrict__ outB,
                                                      int M, int N, int K, int gelu) {
    int lane   = threadIdx.x & 31;
    int wave   = threadIdx.x >> 5;
    int lane16 = lane & 15;
    int kb     = (lane >> 4) << 3;         // +8 K offset for upper half-wave
    int rowBase = blockIdx.y * 128 + (wave & 3) * 32;
    int colBase = blockIdx.x * 128 + (wave >> 2) * 64;

    const unsigned short* arow0 = A + (size_t)(rowBase + lane16) * K;
    const unsigned short* arow1 = A + (size_t)(rowBase + 16 + lane16) * K;
    const unsigned short* wrow0 = W + (size_t)(colBase + lane16) * K;
    const unsigned short* wrow1 = W + (size_t)(colBase + 16 + lane16) * K;
    const unsigned short* wrow2 = W + (size_t)(colBase + 32 + lane16) * K;
    const unsigned short* wrow3 = W + (size_t)(colBase + 48 + lane16) * K;

    v8f acc[2][4];
#pragma unroll
    for (int r = 0; r < 2; ++r)
#pragma unroll
        for (int j = 0; j < 4; ++j) acc[r][j] = (v8f){};

    for (int kk = 0; kk < K; kk += 32) {
        int k0 = kk + kb;
        BFrag a0, a1, b0, b1, b2, b3;
        a0.u4[0] = *(const uint4*)(arow0 + k0);  a0.u4[1] = *(const uint4*)(arow0 + k0 + 16);
        a1.u4[0] = *(const uint4*)(arow1 + k0);  a1.u4[1] = *(const uint4*)(arow1 + k0 + 16);
        b0.u4[0] = *(const uint4*)(wrow0 + k0);  b0.u4[1] = *(const uint4*)(wrow0 + k0 + 16);
        b1.u4[0] = *(const uint4*)(wrow1 + k0);  b1.u4[1] = *(const uint4*)(wrow1 + k0 + 16);
        b2.u4[0] = *(const uint4*)(wrow2 + k0);  b2.u4[1] = *(const uint4*)(wrow2 + k0 + 16);
        b3.u4[0] = *(const uint4*)(wrow3 + k0);  b3.u4[1] = *(const uint4*)(wrow3 + k0 + 16);
        acc[0][0] = wmma_bf16(a0.v, b0.v, acc[0][0]);
        acc[0][1] = wmma_bf16(a0.v, b1.v, acc[0][1]);
        acc[0][2] = wmma_bf16(a0.v, b2.v, acc[0][2]);
        acc[0][3] = wmma_bf16(a0.v, b3.v, acc[0][3]);
        acc[1][0] = wmma_bf16(a1.v, b0.v, acc[1][0]);
        acc[1][1] = wmma_bf16(a1.v, b1.v, acc[1][1]);
        acc[1][2] = wmma_bf16(a1.v, b2.v, acc[1][2]);
        acc[1][3] = wmma_bf16(a1.v, b3.v, acc[1][3]);
    }

#pragma unroll
    for (int r = 0; r < 2; ++r) {
#pragma unroll
        for (int j = 0; j < 4; ++j) {
            int col = colBase + j * 16 + lane16;
            float bs = bias ? bias[col] : 0.0f;
#pragma unroll
            for (int i = 0; i < 8; ++i) {
                int row = rowBase + r * 16 + i + kb;    // C layout: M = i + 8*(lane>>4)
                float v = acc[r][j][i] + bs;
                if (gelu) v = 0.5f * v * (1.0f + erff(v * 0.70710678118654752f));
                size_t o = (size_t)row * N + col;
                if (residual) v += residual[o];
                if (outF) outF[o] = v;
                else      outB[o] = f2bf(v);
            }
        }
    }
}

// ---------------- flash attention over qkv buffer [B,T,3D] (fp32) -> bf16 [B,T,D] ----------------
// grid = (T/64, B*H), block = 128 (4 waves); wave owns 16 query rows, full hd=64.
__global__ __launch_bounds__(128) void attn_fwd_k(const float* __restrict__ qkv,
                                                  unsigned short* __restrict__ aout) {
    __shared__ __align__(16) unsigned short ldsp[4][16 * 32];   // per-wave 16x32 P tile (bf16)

    int lane   = threadIdx.x & 31;
    int wave   = threadIdx.x >> 5;
    int lane16 = lane & 15;
    int kb     = (lane >> 4) << 3;
    int b = blockIdx.y / H_N;
    int h = blockIdx.y % H_N;
    int q0 = blockIdx.x * 64 + wave * 16;
    const int ld = 3 * D_N;
    const float* base = qkv + (size_t)b * T_N * ld + (size_t)h * HD_N;   // +row*ld+k (Q), +D (K), +2D (V)

    // ---- Q fragments (A layout, pre-scaled by 1/sqrt(hd) = 0.125) ----
    BFrag qf[2];
    {
        const float* qp = base + (size_t)(q0 + lane16) * ld;
#pragma unroll
        for (int hh = 0; hh < 2; ++hh) {
            int ko = hh * 32 + kb;
#pragma unroll
            for (int j = 0; j < 8; ++j) {
                qf[hh].s[j]     = f2bf(qp[ko + j] * 0.125f);
                qf[hh].s[8 + j] = f2bf(qp[ko + 16 + j] * 0.125f);
            }
        }
    }

    v8f o0 = {}, o1 = {}, o2 = {}, o3 = {};
    float mrow[8], lrow[8];
#pragma unroll
    for (int i = 0; i < 8; ++i) { mrow[i] = -3.0e38f; lrow[i] = 0.0f; }

    // uniform trip count for the whole block (extra tiles fully masked -> p == 0)
    int nkt = 2 * blockIdx.x + 2;

    for (int kt = 0; kt < nkt; ++kt) {
        int kbase = kt * 32;

        // ---- S = Q * K^T for two 16-key subtiles ----
        v8f s0 = {}, s1 = {};
#pragma unroll
        for (int s = 0; s < 2; ++s) {
            const float* kp = base + (size_t)(kbase + s * 16 + lane16) * ld + D_N;
            BFrag kf0, kf1;
#pragma unroll
            for (int j = 0; j < 8; ++j) {
                kf0.s[j]     = f2bf(kp[kb + j]);
                kf0.s[8 + j] = f2bf(kp[kb + 16 + j]);
                kf1.s[j]     = f2bf(kp[32 + kb + j]);
                kf1.s[8 + j] = f2bf(kp[32 + kb + 16 + j]);
            }
            v8f sc = {};
            sc = wmma_bf16(qf[0].v, kf0.v, sc);
            sc = wmma_bf16(qf[1].v, kf1.v, sc);
            if (s == 0) s0 = sc; else s1 = sc;
        }

        // ---- causal mask + online softmax (rows live across 16-lane groups) ----
#pragma unroll
        for (int i = 0; i < 8; ++i) {
            int qi  = q0 + i + kb;
            int kj0 = kbase + lane16;
            int kj1 = kbase + 16 + lane16;
            float e0 = (kj0 <= qi) ? s0[i] : -3.0e38f;
            float e1 = (kj1 <= qi) ? s1[i] : -3.0e38f;
            float mx = fmaxf(e0, e1);
            for (int m = 1; m < 16; m <<= 1) mx = fmaxf(mx, __shfl_xor(mx, m, 32));
            float mnew  = fmaxf(mrow[i], mx);
            float alpha = __expf(mrow[i] - mnew);
            float p0 = __expf(e0 - mnew);
            float p1 = __expf(e1 - mnew);
            float ps = p0 + p1;
            for (int m = 1; m < 16; m <<= 1) ps += __shfl_xor(ps, m, 32);
            lrow[i] = lrow[i] * alpha + ps;
            mrow[i] = mnew;
            o0[i] *= alpha; o1[i] *= alpha; o2[i] *= alpha; o3[i] *= alpha;
            ldsp[wave][(i + kb) * 32 + lane16]      = f2bf(p0);
            ldsp[wave][(i + kb) * 32 + 16 + lane16] = f2bf(p1);
        }
        __syncthreads();

        // ---- P fragment (A layout) from LDS ----
        BFrag pf;
        pf.u4[0] = *(const uint4*)&ldsp[wave][lane16 * 32 + kb];
        pf.u4[1] = *(const uint4*)&ldsp[wave][lane16 * 32 + kb + 16];

        // ---- O += P * V (V gathered column-major, stride ld) ----
#pragma unroll
        for (int n = 0; n < 4; ++n) {
            int ncol = n * 16 + lane16;
            BFrag vf;
#pragma unroll
            for (int e = 0; e < 16; ++e) {
                int k = ((e >> 3) * 16) + kb + (e & 7);
                vf.s[e] = f2bf(base[(size_t)(kbase + k) * ld + 2 * D_N + ncol]);
            }
            v8f* op = (n == 0) ? &o0 : (n == 1) ? &o1 : (n == 2) ? &o2 : &o3;
            *op = wmma_bf16(pf.v, vf.v, *op);
        }
        __syncthreads();
    }

    // ---- normalize and store bf16 ----
    unsigned short* orow = aout + (size_t)b * T_N * D_N + (size_t)h * HD_N;
#pragma unroll
    for (int n = 0; n < 4; ++n) {
        const v8f& o = (n == 0) ? o0 : (n == 1) ? o1 : (n == 2) ? o2 : o3;
#pragma unroll
        for (int i = 0; i < 8; ++i) {
            int t = q0 + i + kb;
            orow[(size_t)t * D_N + n * 16 + lane16] = f2bf(o[i] / lrow[i]);
        }
    }
}

// ---------------- per-row NLL from logits ----------------
__global__ __launch_bounds__(256) void nll_k(const float* __restrict__ logits,
                                             const int* __restrict__ tgt,
                                             float* __restrict__ nll) {
    __shared__ float red[256];
    int row = blockIdx.x;
    const float* lr = logits + (size_t)row * V_N;
    float mx = -3.0e38f;
    for (int c = threadIdx.x; c < V_N; c += 256) mx = fmaxf(mx, lr[c]);
    red[threadIdx.x] = mx; __syncthreads();
    for (int o = 128; o > 0; o >>= 1) { if (threadIdx.x < o) red[threadIdx.x] = fmaxf(red[threadIdx.x], red[threadIdx.x + o]); __syncthreads(); }
    mx = red[0]; __syncthreads();
    float s = 0.0f;
    for (int c = threadIdx.x; c < V_N; c += 256) s += __expf(lr[c] - mx);
    red[threadIdx.x] = s; __syncthreads();
    for (int o = 128; o > 0; o >>= 1) { if (threadIdx.x < o) red[threadIdx.x] += red[threadIdx.x + o]; __syncthreads(); }
    if (threadIdx.x == 0) nll[row] = -(lr[tgt[row]] - mx - logf(red[0]));
}

__global__ __launch_bounds__(256) void mean_k(const float* __restrict__ nll, float* __restrict__ loss) {
    __shared__ float red[256];
    float s = 0.0f;
    for (int i = threadIdx.x; i < BT_N; i += 256) s += nll[i];
    red[threadIdx.x] = s; __syncthreads();
    for (int o = 128; o > 0; o >>= 1) { if (threadIdx.x < o) red[threadIdx.x] += red[threadIdx.x + o]; __syncthreads(); }
    if (threadIdx.x == 0) *loss = red[0] / (float)BT_N;
}

// ---------------- host orchestration ----------------
extern "C" void kernel_launch(void* const* d_in, const int* in_sizes, int n_in,
                              void* d_out, int out_size, void* d_ws, size_t ws_size,
                              hipStream_t stream) {
    (void)in_sizes; (void)n_in; (void)out_size; (void)ws_size;

    const int*   idx     = (const int*)  d_in[0];
    const int*   targets = (const int*)  d_in[1];
    const float* wte     = (const float*)d_in[2];
    const float* wpe     = (const float*)d_in[3];
    const float* ln1_g   = (const float*)d_in[4];
    const float* ln1_b   = (const float*)d_in[5];
    const float* qkv_w   = (const float*)d_in[6];
    const float* qkv_b   = (const float*)d_in[7];
    const float* out_w   = (const float*)d_in[8];
    const float* out_b   = (const float*)d_in[9];
    const float* ln2_g   = (const float*)d_in[10];
    const float* ln2_b   = (const float*)d_in[11];
    const float* mlp_w1  = (const float*)d_in[12];
    const float* mlp_b1  = (const float*)d_in[13];
    const float* mlp_w2  = (const float*)d_in[14];
    const float* mlp_b2  = (const float*)d_in[15];
    const float* lnf_g   = (const float*)d_in[16];
    const float* lnf_b   = (const float*)d_in[17];
    const float* head_w  = (const float*)d_in[18];

    char* ws = (char*)d_ws;
    size_t off = 0;
    auto alloc = [&](size_t bytes) { size_t o = off; off += (bytes + 255) & ~(size_t)255; return o; };

    // per-layer bf16 weight slabs (recycled every layer)
    unsigned short* wq = (unsigned short*)(ws + alloc((size_t)3 * D_N * D_N * 2));
    unsigned short* wo = (unsigned short*)(ws + alloc((size_t)D_N * D_N * 2));
    unsigned short* w1 = (unsigned short*)(ws + alloc((size_t)4 * D_N * D_N * 2));
    unsigned short* w2 = (unsigned short*)(ws + alloc((size_t)4 * D_N * D_N * 2));
    // activations
    float*          x    = (float*)         (ws + alloc((size_t)BT_N * D_N * 4));
    unsigned short* hbf  = (unsigned short*)(ws + alloc((size_t)BT_N * D_N * 2));
    size_t qkv_off = alloc((size_t)BT_N * 3 * D_N * 4);
    float*          qkvf = (float*)(ws + qkv_off);
    unsigned short* abf  = (unsigned short*)(ws + alloc((size_t)BT_N * D_N * 2));
    unsigned short* mbf  = (unsigned short*)(ws + alloc((size_t)BT_N * 4 * D_N * 2));
    float*          nllp = (float*)(ws + alloc((size_t)BT_N * 4));
    // head weights (65.5MB bf16) alias the dead qkv+attn+mlp activation region (92MB)
    unsigned short* whead = (unsigned short*)(ws + qkv_off);

    float* logits = (float*)d_out;
    float* lossp  = (float*)d_out + (size_t)BT_N * V_N;

    // embeddings
    embed_k<<<BT_N, 256, 0, stream>>>(idx, wte, wpe, x);

    for (int l = 0; l < L_N; ++l) {
        f32_to_bf16_k<<<2048, 256, 0, stream>>>(qkv_w  + (size_t)l * 3 * D_N * D_N, wq, (size_t)3 * D_N * D_N);
        f32_to_bf16_k<<<1024, 256, 0, stream>>>(out_w  + (size_t)l * D_N * D_N,     wo, (size_t)D_N * D_N);
        f32_to_bf16_k<<<2048, 256, 0, stream>>>(mlp_w1 + (size_t)l * 4 * D_N * D_N, w1, (size_t)4 * D_N * D_N);
        f32_to_bf16_k<<<2048, 256, 0, stream>>>(mlp_w2 + (size_t)l * 4 * D_N * D_N, w2, (size_t)4 * D_N * D_N);

        // h = LN1(x)  (bf16)
        layernorm_bf16_k<<<BT_N, 256, 0, stream>>>(x, ln1_g + (size_t)l * D_N, ln1_b + (size_t)l * D_N, hbf);
        // qkv = h @ qkv_w^T + b   [BT, 3D] fp32
        gemm_bf16_wmma<<<dim3(3 * D_N / 128, BT_N / 128), 256, 0, stream>>>(
            hbf, wq, qkv_b + (size_t)l * 3 * D_N, nullptr, qkvf, nullptr, BT_N, 3 * D_N, D_N, 0);
        // flash attention -> abf (bf16 [BT, D])
        attn_fwd_k<<<dim3(T_N / 64, B_N * H_N), 128, 0, stream>>>(qkvf, abf);
        // x = x + abf @ out_w^T + b
        gemm_bf16_wmma<<<dim3(D_N / 128, BT_N / 128), 256, 0, stream>>>(
            abf, wo, out_b + (size_t)l * D_N, x, x, nullptr, BT_N, D_N, D_N, 0);
        // h2 = LN2(x)
        layernorm_bf16_k<<<BT_N, 256, 0, stream>>>(x, ln2_g + (size_t)l * D_N, ln2_b + (size_t)l * D_N, hbf);
        // m = gelu(h2 @ w1^T + b1)  -> bf16
        gemm_bf16_wmma<<<dim3(4 * D_N / 128, BT_N / 128), 256, 0, stream>>>(
            hbf, w1, mlp_b1 + (size_t)l * 4 * D_N, nullptr, nullptr, mbf, BT_N, 4 * D_N, D_N, 1);
        // x = x + m @ w2^T + b2
        gemm_bf16_wmma<<<dim3(D_N / 128, BT_N / 128), 256, 0, stream>>>(
            mbf, w2, mlp_b2 + (size_t)l * D_N, x, x, nullptr, BT_N, D_N, 4 * D_N, 0);
    }

    // final LN + vocab head
    layernorm_bf16_k<<<BT_N, 256, 0, stream>>>(x, lnf_g, lnf_b, hbf);
    f32_to_bf16_k<<<8192, 256, 0, stream>>>(head_w, whead, (size_t)V_N * D_N);
    gemm_bf16_wmma<<<dim3(V_N / 128, BT_N / 128), 256, 0, stream>>>(
        hbf, whead, nullptr, nullptr, logits, nullptr, BT_N, V_N, D_N, 0);

    // loss
    nll_k<<<BT_N, 256, 0, stream>>>(logits, targets, nllp);
    mean_k<<<1, 256, 0, stream>>>(nllp, lossp);
}
